// DagSampler_2963527434283
// MI455X (gfx1250) — matrix operand order repaired and checked by reading, not compiled, not run on hardware
//
#include <hip/hip_runtime.h>

typedef __attribute__((ext_vector_type(16))) _Float16 v16h;
typedef __attribute__((ext_vector_type(8)))  _Float16 v8h;
typedef __attribute__((ext_vector_type(8)))  float    v8f;

#define LDK 40   // LDS row stride in halves (80B: keeps 16B alignment, spreads banks)

// ---------------------------------------------------------------- zero counters
__global__ void zero_counters(unsigned* __restrict__ c, int m) {
    int t = blockIdx.x * blockDim.x + threadIdx.x;
    if (t < m) c[t] = 0u;
}

// ---------------------------------------------------------------- prep: dag matrix + sizes
// dag[i,j] = cpdag[i,j]*(1-cpdag[j,i]); counts: dag edges, lower-tri udg edges.
__global__ void prep_kernel(const float* __restrict__ c, float* __restrict__ dag,
                            unsigned* __restrict__ cnt, int n, int S) {
    __shared__ float ct[16][17];
    int tx = threadIdx.x, ty = threadIdx.y;
    int i0 = blockIdx.y * 16, j0 = blockIdx.x * 16;
    ct[ty][tx] = c[(size_t)(j0 + ty) * n + (i0 + tx)];
    __syncthreads();
    int i = i0 + ty, j = j0 + tx;
    float cij = c[(size_t)i * n + j];
    float cji = ct[tx][ty];
    float dagij = cij * (1.0f - cji);
    float udg   = cij * cji;
    dag[(size_t)i * n + j] = dagij;
    int dc = (dagij > 0.5f) ? 1 : 0;
    int uc = (i > j && udg > 0.5f) ? 1 : 0;
    int lane = (ty * 16 + tx) & 31;
#pragma unroll
    for (int off = 16; off; off >>= 1) {
        dc += __shfl_xor(dc, off, 32);
        uc += __shfl_xor(uc, off, 32);
    }
    if (lane == 0) {
        atomicAdd(&cnt[S + 1], (unsigned)dc);
        atomicAdd(&cnt[S + 2], (unsigned)uc);
    }
}

// ---------------------------------------------------------------- build full[s,i,j]
// full = dag_ij + [j>i]*udg*(1 - s(u_ji)) + [i>j]*udg*s(u_ij)
// with s(u) = sigmoid(0.5 + logit(u)) = u / (u + e^{-0.5}*(1-u))   (no transcendentals)
// 32x32 tile per block (32,8 threads, 4 rows each) -> 128B coalesced rows.
__global__ __launch_bounds__(256)
void build_full(const float* __restrict__ c, const float* __restrict__ u,
                float* __restrict__ full, int n) {
    __shared__ float ct[32][33];
    __shared__ float ut[32][33];
    int tx = threadIdx.x;            // 0..31
    int ty = threadIdx.y;            // 0..7
    int s = blockIdx.z;
    int i0 = blockIdx.y * 32, j0 = blockIdx.x * 32;
    size_t base = (size_t)s * n * n;
    // stage transposed tiles (coalesced 128B rows)
#pragma unroll
    for (int rr = 0; rr < 4; ++rr) {
        int row = ty + rr * 8;
        ct[row][tx] = c[(size_t)(j0 + row) * n + (i0 + tx)];
        ut[row][tx] = u[base + (size_t)(j0 + row) * n + (i0 + tx)];
    }
    __syncthreads();
    const float C = 0.60653065971263342f; // e^{-0.5}
#pragma unroll
    for (int rr = 0; rr < 4; ++rr) {
        int li = ty + rr * 8;
        int i = i0 + li, j = j0 + tx;
        float cij = c[(size_t)i * n + j];
        float uij = u[base + (size_t)i * n + j];
        float cji = ct[tx][li];
        float uji = ut[tx][li];
        float udg   = cij * cji;
        float dagij = cij * (1.0f - cji);
        float sij = uij / (uij + C * (1.0f - uij));
        float sji = uji / (uji + C * (1.0f - uji));
        float val = dagij;
        if (j > i) val += udg * (1.0f - sji);
        if (i > j) val += udg * sij;
        full[base + (size_t)i * n + j] = val;
    }
}

// ---------------------------------------------------------------- WMMA gemm + immorality count
// One block: 128x128 output tile of gg = M*M^T (upper-tri blocks only), then
// counts (M[i,k]+M[k,i]-gg[i,k] < 0) over i<k, accumulated into counters[s].
// Register-double-buffered global->LDS staging (f32 read from L2, f16 in LDS).
__global__ __launch_bounds__(256)
void gemm_count(const float* __restrict__ full, const float* __restrict__ dagbuf,
                unsigned* __restrict__ counters, int n, int S) {
    __shared__ __attribute__((aligned(16))) _Float16 lA[128 * LDK];
    __shared__ __attribute__((aligned(16))) _Float16 lB[128 * LDK];

    int s = blockIdx.y;
    const float* __restrict__ m = (s < S) ? (full + (size_t)s * n * n) : dagbuf;

    // decode (bi,bk) with bi<=bk from linear pair index
    int nb = n / 128;
    int p = blockIdx.x;
    int bi = 0;
    while (p >= nb - bi) { p -= (nb - bi); ++bi; }
    int bk = bi + p;
    int i0 = bi * 128, k0 = bk * 128;

    int tid = threadIdx.x;
    int lane = tid & 31;
    int w = tid >> 5;          // 8 waves
    int wi = w & 3;            // 4 row groups of 32 rows
    int wk = w >> 2;           // 2 col groups of 64 cols

    v8f acc[2][4] = {};        // 2 row-tiles x 4 col-tiles of 16x16

    int cg = tid & 7;          // col-group of 4 floats within 32-wide panel
    int r0 = tid >> 3;         // 0..31

    const float* __restrict__ mA = m + (size_t)i0 * n;
    const float* __restrict__ mB = m + (size_t)k0 * n;   // may equal mA (diag block)

    // prologue: fetch K-chunk 0 into registers (8 independent 128b loads)
    float4 ra[4], rb[4];
#pragma unroll
    for (int p4 = 0; p4 < 4; ++p4) {
        size_t roff = (size_t)(p4 * 32 + r0) * n + cg * 4;
        ra[p4] = *(const float4*)&mA[roff];
        rb[p4] = *(const float4*)&mB[roff];
    }

    for (int kb = 0; kb < n; kb += 32) {
        // stage current chunk to LDS as f16
#pragma unroll
        for (int p4 = 0; p4 < 4; ++p4) {
            int r = p4 * 32 + r0;
            _Float16* da = &lA[r * LDK + cg * 4];
            da[0] = (_Float16)ra[p4].x; da[1] = (_Float16)ra[p4].y;
            da[2] = (_Float16)ra[p4].z; da[3] = (_Float16)ra[p4].w;
            _Float16* db = &lB[r * LDK + cg * 4];
            db[0] = (_Float16)rb[p4].x; db[1] = (_Float16)rb[p4].y;
            db[2] = (_Float16)rb[p4].z; db[3] = (_Float16)rb[p4].w;
        }
        __syncthreads();

        // prefetch next chunk while WMMAs run (loads only touch registers)
        if (kb + 32 < n) {
            size_t koff = (size_t)(kb + 32) + cg * 4;
#pragma unroll
            for (int p4 = 0; p4 < 4; ++p4) {
                size_t roff = (size_t)(p4 * 32 + r0) * n + koff;
                ra[p4] = *(const float4*)&mA[roff];
                rb[p4] = *(const float4*)&mB[roff];
            }
        }

        // A fragments: lane<16 holds row (lane), K {0..7,16..23}; lane>=16 K {8..15,24..31}
        v16h af[2];
        {
            int r = lane & 15;
            int koff = (lane >> 4) * 8;
#pragma unroll
            for (int rt = 0; rt < 2; ++rt) {
                const _Float16* pa = &lA[(wi * 32 + rt * 16 + r) * LDK + koff];
                v8h lo = *(const v8h*)pa;
                v8h hi = *(const v8h*)(pa + 16);
                v16h f;
#pragma unroll
                for (int q = 0; q < 8; ++q) { f[q] = lo[q]; f[8 + q] = hi[q]; }
                af[rt] = f;
            }
        }
        // B fragments: lane<16 holds col (lane), K {0..7,16..23}; lane>=16 K {8..15,24..31}
        v16h bf[4];
        {
            int cI = lane & 15;
            int koff = (lane >> 4) * 8;
#pragma unroll
            for (int ct = 0; ct < 4; ++ct) {
                const _Float16* pb = &lB[(wk * 64 + ct * 16 + cI) * LDK + koff];
                v8h lo = *(const v8h*)pb;
                v8h hi = *(const v8h*)(pb + 16);
                v16h f;
#pragma unroll
                for (int q = 0; q < 8; ++q) { f[q] = lo[q]; f[8 + q] = hi[q]; }
                bf[ct] = f;
            }
        }
#pragma unroll
        for (int rt = 0; rt < 2; ++rt)
#pragma unroll
            for (int ct = 0; ct < 4; ++ct)
                acc[rt][ct] = __builtin_amdgcn_wmma_f32_16x16x32_f16(
                    false, af[rt], false, bf[ct], (short)0, acc[rt][ct], false, false);
        __syncthreads();
    }

    // count a = m[i,k] + m[k,i] - gg < 0 over global i<k
    int cnt = 0;
    int igb = i0 + wi * 32 + ((lane >> 4) << 3);   // + rt*16 + e
    int kgb = k0 + wk * 64 + (lane & 15);          // + ct*16
#pragma unroll
    for (int rt = 0; rt < 2; ++rt) {
#pragma unroll
        for (int ct = 0; ct < 4; ++ct) {
#pragma unroll
            for (int e = 0; e < 8; ++e) {
                int ig = igb + rt * 16 + e;
                int kg = kgb + ct * 16;
                float gg = acc[rt][ct][e];
                float a = m[(size_t)ig * n + kg] + m[(size_t)kg * n + ig] - gg;
                cnt += (ig < kg && a < 0.0f) ? 1 : 0;
            }
        }
    }
#pragma unroll
    for (int off = 16; off; off >>= 1) cnt += __shfl_xor(cnt, off, 32);
    if (lane == 0) atomicAdd(&counters[s], (unsigned)cnt);
}

// ---------------------------------------------------------------- finalize scalars
__global__ void finalize(const unsigned* __restrict__ counters, float* __restrict__ out,
                         int n, int S) {
    __shared__ unsigned match;
    int t = threadIdx.x;
    if (t == 0) match = 0u;
    __syncthreads();
    size_t offW = (size_t)S * n * n;
    unsigned dag_imm = counters[S];
    if (t < S) {
        float Sf = (float)S;
        float norm = (1.0f / (1.0f + Sf)) * ((Sf + 1.0f) / Sf);
        out[offW + t] = norm;                       // weights (w=1 each)
        // s_cycles == dag_cycles always (all diagonals exactly 0 -> trace(exp)=n)
        if (counters[t] == dag_imm) atomicAdd(&match, 1u);
    }
    __syncthreads();
    if (t == 0) {
        out[offW + S]     = (float)match / (float)S;   // avg_correct
        out[offW + S + 1] = (float)counters[S + 1];    // dag_size
        out[offW + S + 2] = (float)counters[S + 2];    // udg_size
    }
}

// ---------------------------------------------------------------- launch
extern "C" void kernel_launch(void* const* d_in, const int* in_sizes, int n_in,
                              void* d_out, int out_size, void* d_ws, size_t ws_size,
                              hipStream_t stream) {
    int nn = in_sizes[0];            // n*n
    int n = 1;
    while ((long long)n * n < (long long)nn) ++n;     // n = 1024
    int S = in_sizes[1] / nn;                          // 64

    const float* cpdag = (const float*)d_in[0];
    const float* u     = (const float*)d_in[1];
    float* full = (float*)d_out;

    float* dagbuf = (float*)d_ws;                              // n*n f32
    unsigned* counters = (unsigned*)(dagbuf + (size_t)nn);     // S+3 uints

    zero_counters<<<1, 128, 0, stream>>>(counters, S + 3);

    prep_kernel<<<dim3(n / 16, n / 16), dim3(16, 16), 0, stream>>>(cpdag, dagbuf, counters, n, S);
    build_full<<<dim3(n / 32, n / 32, S), dim3(32, 8), 0, stream>>>(cpdag, u, full, n);

    int nb = n / 128;
    int npairs = nb * (nb + 1) / 2;                  // upper-tri blocks only (a symmetric)
    gemm_count<<<dim3(npairs, S + 1), 256, 0, stream>>>(full, dagbuf, counters, n, S);

    finalize<<<1, 256, 0, stream>>>(counters, full, n, S);
}
